// PointNet2Feat_17291538333862
// MI455X (gfx1250) — matrix operand order repaired and checked by reading, not compiled, not run on hardware
//
#include <hip/hip_runtime.h>
#include <hip/hip_bf16.h>

typedef __attribute__((ext_vector_type(16))) _Float16 v16h;
typedef __attribute__((ext_vector_type(8)))  float    v8f;

#define B_   16
#define N_   8192
#define S1_  512
#define S2_  128
#define NS_  64

// ---------------- WMMA fragment helpers (CDNA5 16x16x32 f16, wave32) -------
// A (16x32, 16-bit): lane<16 -> kb=0, lane>=16 -> kb=8; elems 0..7 = K kb..kb+7,
// elems 8..15 = K kb+16..kb+23 (ISA 7.12.2).
__device__ __forceinline__ v16h frag_a_lds(const _Float16* As, int stride,
                                           int lane, int kbase) {
  const int m  = lane & 15;
  const int kb = kbase + ((lane & 16) ? 8 : 0);
  v16h a;
#pragma unroll
  for (int e = 0; e < 8; ++e) a[e] = As[m * stride + kb + e];
#pragma unroll
  for (int e = 0; e < 8; ++e) a[8 + e] = As[m * stride + kb + 16 + e];
  return a;
}

// B fragment from pre-packed weights: Wpk[((nt*KT + kt)*32 + lane)*16 .. +15]
// is exactly this lane's 16 f16 elements -> one contiguous 32B load.
__device__ __forceinline__ v16h frag_b_pk(const _Float16* __restrict__ Wpk,
                                          int nt, int KT, int kt, int lane) {
  return *(const v16h*)(Wpk + (((size_t)nt * KT + kt) * 32 + lane) * 16);
}

__device__ __forceinline__ v8f wmma_f16(v16h a, v16h b, v8f c) {
  return __builtin_amdgcn_wmma_f32_16x16x32_f16(false, a, false, b,
                                                (short)0, c, false, false);
}

// ---------------- Weight pre-pack: f32 [Cout][Cin] -> f16 B-fragment layout -
// One thread per (nt, kt, lane); writes 16 f16 (zero-padded past Cin/Cout).
// B (32x16, 16-bit): lane = N column; elems 0..7 = K kb..kb+7, elems 8..15 =
// K kb+16..kb+23 with kb = kt*32 (+8 for lanes 16-31).
__global__ __launch_bounds__(256) void k_prepack(const float* __restrict__ W,
                                                 int Cout, int Cin, int NT,
                                                 int KT,
                                                 _Float16* __restrict__ out) {
  const int t = blockIdx.x * 256 + threadIdx.x;  // (nt*KT + kt)*32 + lane
  const int total = NT * KT * 32;
  if (t >= total) return;
  const int lane = t & 31;
  const int ktn  = t >> 5;
  const int kt = ktn % KT, nt = ktn / KT;
  const int n  = nt * 16 + (lane & 15);
  const int kb = kt * 32 + ((lane & 16) ? 8 : 0);
  _Float16* dst = out + (size_t)t * 16;
#pragma unroll
  for (int e = 0; e < 16; ++e) {
    const int k = kb + (e < 8 ? e : e + 8);
    dst[e] = (k < Cin && n < Cout) ? (_Float16)W[(size_t)n * Cin + k]
                                   : (_Float16)0.0f;
  }
}

// ---------------- Farthest point sampling (one block per batch) ------------
template <int NPTS, int NPOINT, int THREADS>
__global__ __launch_bounds__(THREADS) void k_fps(const float* __restrict__ xyz,
                                                 float* __restrict__ nxyz,
                                                 int* __restrict__ fidx) {
  __shared__ float mind[NPTS];
  __shared__ float rv[THREADS];
  __shared__ int   ri[THREADS];
  __shared__ int   s_last;
  const int b = blockIdx.x, tid = threadIdx.x;
  const float* px = xyz + (size_t)b * NPTS * 3;
  for (int j = tid; j < NPTS; j += THREADS) mind[j] = 1e10f;
  if (tid == 0) s_last = 0;
  __syncthreads();
  for (int it = 0; it < NPOINT; ++it) {
    const int last = s_last;
    const float lx = px[last * 3 + 0];
    const float ly = px[last * 3 + 1];
    const float lz = px[last * 3 + 2];
    if (tid == 0) {
      fidx[b * NPOINT + it] = last;
      nxyz[((size_t)b * NPOINT + it) * 3 + 0] = lx;
      nxyz[((size_t)b * NPOINT + it) * 3 + 1] = ly;
      nxyz[((size_t)b * NPOINT + it) * 3 + 2] = lz;
    }
    float bv = -1.0f;
    int   bi = 0;
    for (int j = tid; j < NPTS; j += THREADS) {
      const float dx = px[j * 3 + 0] - lx;
      const float dy = px[j * 3 + 1] - ly;
      const float dz = px[j * 3 + 2] - lz;
      const float d  = dx * dx + dy * dy + dz * dz;
      const float m  = fminf(mind[j], d);
      mind[j] = m;
      if (m > bv) { bv = m; bi = j; }
    }
    rv[tid] = bv; ri[tid] = bi;
    __syncthreads();
    for (int s = THREADS >> 1; s > 0; s >>= 1) {
      if (tid < s) {
        const float v2 = rv[tid + s]; const int i2 = ri[tid + s];
        if (v2 > rv[tid] || (v2 == rv[tid] && i2 < ri[tid])) {
          rv[tid] = v2; ri[tid] = i2;
        }
      }
      __syncthreads();
    }
    if (tid == 0) s_last = ri[0];
    __syncthreads();
  }
}

// ---------------- Ball query: one wave32 per center ------------------------
template <int NPTS>
__global__ __launch_bounds__(256) void k_ballquery(
    const float* __restrict__ xyz, const float* __restrict__ cxyz,
    int* __restrict__ idx, int S, float r2) {
  const int wave = threadIdx.x >> 5, lane = threadIdx.x & 31;
  const int c = blockIdx.x * 8 + wave;   // global center id over B*S
  if (c >= B_ * S) return;
  const int b = c / S;
  const float qx = cxyz[c * 3 + 0];
  const float qy = cxyz[c * 3 + 1];
  const float qz = cxyz[c * 3 + 2];
  const float* px = xyz + (size_t)b * NPTS * 3;
  int* out = idx + (size_t)c * NS_;
  int cnt = 0, first = NPTS - 1;
  for (int base = 0; base < NPTS && cnt < NS_; base += 32) {
    const int j = base + lane;
    bool pred = false;
    if (j < NPTS) {
      const float dx = px[j * 3 + 0] - qx;
      const float dy = px[j * 3 + 1] - qy;
      const float dz = px[j * 3 + 2] - qz;
      pred = (dx * dx + dy * dy + dz * dz) < r2;
    }
    const unsigned m = (unsigned)__ballot((int)pred);
    if (cnt == 0 && m) first = base + __ffs(m) - 1;
    const int pos = cnt + __popc(m & ((1u << lane) - 1u));
    if (pred && pos < NS_) out[pos] = j;
    cnt += __popc(m);
  }
  if (cnt > NS_) cnt = NS_;
  for (int s = cnt + lane; s < NS_; s += 32) out[s] = first;
}

// ---------------- SA1 fused: gather + (3->64->128) MLP + maxpool -----------
__global__ __launch_bounds__(256) void k_sa1(
    const float* __restrict__ xyz, const float* __restrict__ nxyz1,
    const int* __restrict__ idx1, const _Float16* __restrict__ W1apk,
    const float* __restrict__ b1a, const _Float16* __restrict__ W1bpk,
    const float* __restrict__ b1b, _Float16* __restrict__ feat1) {
  __shared__ _Float16 As[64 * 32];
  __shared__ _Float16 Mid[64 * 64];
  __shared__ int      pmax[128];
  const int g = blockIdx.x;           // b*S1 + s
  const int b = g / S1_;
  const int tid = threadIdx.x, wave = tid >> 5, lane = tid & 31;
  const int* gi = idx1 + (size_t)g * NS_;
  const float cx = nxyz1[g * 3 + 0];
  const float cy = nxyz1[g * 3 + 1];
  const float cz = nxyz1[g * 3 + 2];
  for (int e = tid; e < 64 * 32; e += 256) {
    const int rrow = e >> 5, cch = e & 31;
    float v = 0.0f;
    if (cch < 3) {
      const int p = gi[rrow];
      v = xyz[((size_t)b * N_ + p) * 3 + cch] -
          (cch == 0 ? cx : (cch == 1 ? cy : cz));
    }
    As[e] = (_Float16)v;
  }
  for (int c = tid; c < 128; c += 256) pmax[c] = 0;
  __syncthreads();
  // layer a: 4 M-tiles x 4 N-tiles, K padded 3->32 (KT=1)
  for (int job = wave; job < 16; job += 8) {
    const int mt = job >> 2, nt = job & 3;
    v8f acc = {};
    const v16h a = frag_a_lds(As + mt * 16 * 32, 32, lane, 0);
    const v16h w = frag_b_pk(W1apk, nt, 1, 0, lane);
    acc = wmma_f16(a, w, acc);
    const int row0 = (lane & 16) ? 8 : 0;
    const int col  = nt * 16 + (lane & 15);
    const float bias = b1a[col];
#pragma unroll
    for (int r = 0; r < 8; ++r) {
      float v = acc[r] + bias; v = v > 0.0f ? v : 0.0f;
      Mid[(mt * 16 + row0 + r) * 64 + col] = (_Float16)v;
    }
  }
  __syncthreads();
  // layer b: 4 M-tiles x 8 N-tiles, K = 64 (KT=2), maxpool over 64 rows
  for (int job = wave; job < 32; job += 8) {
    const int mt = job >> 3, nt = job & 7;
    v8f acc = {};
#pragma unroll
    for (int kt = 0; kt < 2; ++kt) {
      const v16h a = frag_a_lds(Mid + mt * 16 * 64, 64, lane, kt * 32);
      const v16h w = frag_b_pk(W1bpk, nt, 2, kt, lane);
      acc = wmma_f16(a, w, acc);
    }
    const int col = nt * 16 + (lane & 15);
    const float bias = b1b[col];
#pragma unroll
    for (int r = 0; r < 8; ++r) {
      float v = acc[r] + bias; v = v > 0.0f ? v : 0.0f;
      atomicMax(&pmax[col], __float_as_int(v));   // relu >= 0 -> int-bit max ok
    }
  }
  __syncthreads();
  for (int c = tid; c < 128; c += 256)
    feat1[(size_t)g * 128 + c] = (_Float16)__int_as_float(pmax[c]);
}

// ---------------- SA2 fused: gather + (131->128->256) MLP + maxpool --------
__global__ __launch_bounds__(256) void k_sa2(
    const float* __restrict__ xyz1, const _Float16* __restrict__ feat1,
    const float* __restrict__ nxyz2, const int* __restrict__ idx2,
    const _Float16* __restrict__ W2apk, const float* __restrict__ b2a,
    const _Float16* __restrict__ W2bpk, const float* __restrict__ b2b,
    _Float16* __restrict__ feat2) {
  __shared__ _Float16 As[64 * 160];    // 131 channels padded to 160
  __shared__ _Float16 Mid[64 * 128];
  __shared__ int      pmax[256];
  const int g = blockIdx.x;            // b*S2 + s
  const int b = g >> 7;
  const int tid = threadIdx.x, wave = tid >> 5, lane = tid & 31;
  const int* gi = idx2 + (size_t)g * NS_;
  const float cx = nxyz2[g * 3 + 0];
  const float cy = nxyz2[g * 3 + 1];
  const float cz = nxyz2[g * 3 + 2];
  for (int e = tid; e < 64 * 160; e += 256) {
    const int rrow = e / 160, cch = e % 160;
    float v = 0.0f;
    const int p = gi[rrow];
    if (cch < 3)
      v = xyz1[((size_t)b * S1_ + p) * 3 + cch] -
          (cch == 0 ? cx : (cch == 1 ? cy : cz));
    else if (cch < 131)
      v = (float)feat1[((size_t)b * S1_ + p) * 128 + (cch - 3)];
    As[e] = (_Float16)v;
  }
  for (int c = tid; c < 256; c += 256) pmax[c] = 0;
  __syncthreads();
  // layer a: 4 mt x 8 nt, KT=5 (Cin=131 padded 160)
  for (int job = wave; job < 32; job += 8) {
    const int mt = job >> 3, nt = job & 7;
    v8f acc = {};
#pragma unroll
    for (int kt = 0; kt < 5; ++kt) {
      const v16h a = frag_a_lds(As + mt * 16 * 160, 160, lane, kt * 32);
      const v16h w = frag_b_pk(W2apk, nt, 5, kt, lane);
      acc = wmma_f16(a, w, acc);
    }
    const int row0 = (lane & 16) ? 8 : 0;
    const int col  = nt * 16 + (lane & 15);
    const float bias = b2a[col];
#pragma unroll
    for (int r = 0; r < 8; ++r) {
      float v = acc[r] + bias; v = v > 0.0f ? v : 0.0f;
      Mid[(mt * 16 + row0 + r) * 128 + col] = (_Float16)v;
    }
  }
  __syncthreads();
  // layer b: 4 mt x 16 nt, K = 128 (KT=4), maxpool
  for (int job = wave; job < 64; job += 8) {
    const int mt = job >> 4, nt = job & 15;
    v8f acc = {};
#pragma unroll
    for (int kt = 0; kt < 4; ++kt) {
      const v16h a = frag_a_lds(Mid + mt * 16 * 128, 128, lane, kt * 32);
      const v16h w = frag_b_pk(W2bpk, nt, 4, kt, lane);
      acc = wmma_f16(a, w, acc);
    }
    const int col = nt * 16 + (lane & 15);
    const float bias = b2b[col];
#pragma unroll
    for (int r = 0; r < 8; ++r) {
      float v = acc[r] + bias; v = v > 0.0f ? v : 0.0f;
      atomicMax(&pmax[col], __float_as_int(v));
    }
  }
  __syncthreads();
  for (int c = tid; c < 256; c += 256)
    feat2[(size_t)g * 256 + c] = (_Float16)__int_as_float(pmax[c]);
}

// ---------------- SA3 layer a: (259->512), block = one 16-row M-tile -------
__global__ __launch_bounds__(256) void k_sa3a(
    const float* __restrict__ nxyz2, const _Float16* __restrict__ feat2,
    const _Float16* __restrict__ W3apk, const float* __restrict__ b3a,
    _Float16* __restrict__ act3) {
  __shared__ _Float16 As[16 * 288];    // 259 channels padded to 288
  const int b = blockIdx.x >> 3, mt = blockIdx.x & 7;
  const int tid = threadIdx.x, wave = tid >> 5, lane = tid & 31;
  for (int e = tid; e < 16 * 288; e += 256) {
    const int rrow = e / 288, cch = e % 288;
    const int s = mt * 16 + rrow;
    float v = 0.0f;
    if (cch < 3)
      v = nxyz2[((size_t)b * S2_ + s) * 3 + cch];
    else if (cch < 259)
      v = (float)feat2[((size_t)b * S2_ + s) * 256 + (cch - 3)];
    As[e] = (_Float16)v;
  }
  __syncthreads();
  for (int nt = wave; nt < 32; nt += 8) {
    v8f acc = {};
#pragma unroll
    for (int kt = 0; kt < 9; ++kt) {
      const v16h a = frag_a_lds(As, 288, lane, kt * 32);
      const v16h w = frag_b_pk(W3apk, nt, 9, kt, lane);
      acc = wmma_f16(a, w, acc);
    }
    const int row0 = (lane & 16) ? 8 : 0;
    const int col  = nt * 16 + (lane & 15);
    const float bias = b3a[col];
#pragma unroll
    for (int r = 0; r < 8; ++r) {
      float v = acc[r] + bias; v = v > 0.0f ? v : 0.0f;
      const int s = mt * 16 + row0 + r;
      act3[((size_t)b * S2_ + s) * 512 + col] = (_Float16)v;
    }
  }
}

// ---------------- SA3 layer b: (512->1024) + maxpool over 128 points -------
__global__ __launch_bounds__(256) void k_sa3b(
    const _Float16* __restrict__ act3, const _Float16* __restrict__ W3bpk,
    const float* __restrict__ b3b, float* __restrict__ out) {
  __shared__ _Float16 As[16 * 512];
  __shared__ int      pmax[1024];
  const int b = blockIdx.x;
  const int tid = threadIdx.x, wave = tid >> 5, lane = tid & 31;
  for (int c = tid; c < 1024; c += 256) pmax[c] = 0;
  for (int mt = 0; mt < 8; ++mt) {
    __syncthreads();
    for (int e = tid; e < 16 * 512; e += 256) {
      const int rrow = e >> 9, cch = e & 511;
      As[e] = act3[((size_t)b * S2_ + mt * 16 + rrow) * 512 + cch];
    }
    __syncthreads();
    for (int nt = wave; nt < 64; nt += 8) {
      v8f acc = {};
#pragma unroll
      for (int kt = 0; kt < 16; ++kt) {
        const v16h a = frag_a_lds(As, 512, lane, kt * 32);
        const v16h w = frag_b_pk(W3bpk, nt, 16, kt, lane);
        acc = wmma_f16(a, w, acc);
      }
      const int col = nt * 16 + (lane & 15);
      const float bias = b3b[col];
#pragma unroll
      for (int r = 0; r < 8; ++r) {
        float v = acc[r] + bias; v = v > 0.0f ? v : 0.0f;
        atomicMax(&pmax[col], __float_as_int(v));
      }
    }
  }
  __syncthreads();
  for (int c = tid; c < 1024; c += 256)
    out[(size_t)b * 1024 + c] = __int_as_float(pmax[c]);
}

// ---------------------------------------------------------------------------
extern "C" void kernel_launch(void* const* d_in, const int* in_sizes, int n_in,
                              void* d_out, int out_size, void* d_ws,
                              size_t ws_size, hipStream_t stream) {
  (void)in_sizes; (void)n_in; (void)out_size; (void)ws_size;
  const float* x   = (const float*)d_in[0];
  const float* W1a = (const float*)d_in[1];
  const float* b1a = (const float*)d_in[2];
  const float* W1b = (const float*)d_in[3];
  const float* b1b = (const float*)d_in[4];
  const float* W2a = (const float*)d_in[5];
  const float* b2a = (const float*)d_in[6];
  const float* W2b = (const float*)d_in[7];
  const float* b2b = (const float*)d_in[8];
  const float* W3a = (const float*)d_in[9];
  const float* b3a = (const float*)d_in[10];
  const float* W3b = (const float*)d_in[11];
  const float* b3b = (const float*)d_in[12];

  uint8_t* ws = (uint8_t*)d_ws;
  float*    nxyz1 = (float*)(ws + 0);            // 16*512*3*4  = 98304
  int*      fidx1 = (int*)(ws + 98304);          // 16*512*4    = 32768
  int*      idx1  = (int*)(ws + 131072);         // 16*512*64*4 = 2097152
  _Float16* feat1 = (_Float16*)(ws + 2228224);   // 16*512*128*2= 2097152
  float*    nxyz2 = (float*)(ws + 4325376);      // 16*128*3*4  = 24576
  int*      fidx2 = (int*)(ws + 4349952);        // 16*128*4    = 8192
  int*      idx2  = (int*)(ws + 4358144);        // 16*128*64*4 = 524288
  _Float16* feat2 = (_Float16*)(ws + 4882432);   // 16*128*256*2= 1048576
  _Float16* act3  = (_Float16*)(ws + 5931008);   // 16*128*512*2= 2097152
  // pre-packed weights (f16, B-fragment layout), NT*KT*512 halves each:
  _Float16* W1apk = (_Float16*)(ws + 8028160);   //  4*1 *1024B =   4096
  _Float16* W1bpk = (_Float16*)(ws + 8032256);   //  8*2        =  16384
  _Float16* W2apk = (_Float16*)(ws + 8048640);   //  8*5        =  40960
  _Float16* W2bpk = (_Float16*)(ws + 8089600);   // 16*4        =  65536
  _Float16* W3apk = (_Float16*)(ws + 8155136);   // 32*9        = 294912
  _Float16* W3bpk = (_Float16*)(ws + 8450048);   // 64*16       = 1048576
  float*    out   = (float*)d_out;               // [16,1024]

  // ---- weight pre-pack (tiny; overlaps FPS on other CUs) ----
  k_prepack<<<1,   256, 0, stream>>>(W1a,   64,   3,  4,  1, W1apk);
  k_prepack<<<2,   256, 0, stream>>>(W1b,  128,  64,  8,  2, W1bpk);
  k_prepack<<<5,   256, 0, stream>>>(W2a,  128, 131,  8,  5, W2apk);
  k_prepack<<<8,   256, 0, stream>>>(W2b,  256, 128, 16,  4, W2bpk);
  k_prepack<<<36,  256, 0, stream>>>(W3a,  512, 259, 32,  9, W3apk);
  k_prepack<<<128, 256, 0, stream>>>(W3b, 1024, 512, 64, 16, W3bpk);

  // ---- SA1 ----
  k_fps<N_, S1_, 1024><<<B_, 1024, 0, stream>>>(x, nxyz1, fidx1);
  k_ballquery<N_><<<(B_ * S1_) / 8, 256, 0, stream>>>(x, nxyz1, idx1, S1_,
                                                      0.2f * 0.2f);
  k_sa1<<<B_ * S1_, 256, 0, stream>>>(x, nxyz1, idx1, W1apk, b1a, W1bpk, b1b,
                                      feat1);
  // ---- SA2 ----
  k_fps<S1_, S2_, 256><<<B_, 256, 0, stream>>>(nxyz1, nxyz2, fidx2);
  k_ballquery<S1_><<<(B_ * S2_) / 8, 256, 0, stream>>>(nxyz1, nxyz2, idx2, S2_,
                                                       0.4f * 0.4f);
  k_sa2<<<B_ * S2_, 256, 0, stream>>>(nxyz1, feat1, nxyz2, idx2, W2apk, b2a,
                                      W2bpk, b2b, feat2);
  // ---- SA3 (global) ----
  k_sa3a<<<B_ * 8, 256, 0, stream>>>(nxyz2, feat2, W3apk, b3a, act3);
  k_sa3b<<<B_, 256, 0, stream>>>(act3, W3bpk, b3b, out);
}
